// DualBranchSPPF_LSKA_59382217834739
// MI455X (gfx1250) — matrix operand order repaired and checked by reading, not compiled, not run on hardware
//
#include <hip/hip_runtime.h>
#include <hip/hip_bf16.h>

// CDNA5 / gfx1250 implementation.
// 1x1 convs = bf16 WMMA GEMMs (v_wmma_f32_16x16x32_bf16, fp32 acc), register
// tiled 32(Cout) x 64(px) per wave (2 A-frags x 4 B-frags -> 8 WMMA / K-step).
// Pools are LDS-tiled window kernels, depthwise convs are streaming VALU.

typedef __attribute__((ext_vector_type(16))) __bf16 v16bf;
typedef __attribute__((ext_vector_type(8)))  float  v8f;

#define HWD  64
#define PIX  4096           // 64*64 pixels per (batch, channel) plane
#define TPOOL 0.9f

__device__ __forceinline__ float silu_f(float v) {
    // v * sigmoid(v); v_rcp_f32 instead of IEEE divide (activation precision)
    return v * __builtin_amdgcn_rcpf(1.0f + __expf(-v));
}

// A fragment: weight row, lane-half covers K = {h*8..h*8+7, 16+h*8..16+h*8+7}
__device__ __forceinline__ v16bf load_a_frag(const float* __restrict__ wrow) {
    const float4 a0 = *(const float4*)(wrow);
    const float4 a1 = *(const float4*)(wrow + 4);
    const float4 a2 = *(const float4*)(wrow + 16);
    const float4 a3 = *(const float4*)(wrow + 20);
    v16bf A;
    A[0]=(__bf16)a0.x;  A[1]=(__bf16)a0.y;  A[2]=(__bf16)a0.z;  A[3]=(__bf16)a0.w;
    A[4]=(__bf16)a1.x;  A[5]=(__bf16)a1.y;  A[6]=(__bf16)a1.z;  A[7]=(__bf16)a1.w;
    A[8]=(__bf16)a2.x;  A[9]=(__bf16)a2.y;  A[10]=(__bf16)a2.z; A[11]=(__bf16)a2.w;
    A[12]=(__bf16)a3.x; A[13]=(__bf16)a3.y; A[14]=(__bf16)a3.z; A[15]=(__bf16)a3.w;
    return A;
}

// B fragment: lane holds pixel column, K rows {h*8..h*8+7, 16+h*8..16+h*8+7}
__device__ __forceinline__ v16bf load_b_frag(const float* __restrict__ brow) {
    v16bf B;
    #pragma unroll
    for (int s = 0; s < 8; ++s) {
        B[s]     = (__bf16)brow[(long)s * PIX];
        B[8 + s] = (__bf16)brow[(long)(16 + s) * PIX];
    }
    return B;
}

// B fragment with elementwise gating multiply (y * a) fused before conversion
__device__ __forceinline__ v16bf load_b_frag_mul(const float* __restrict__ brow,
                                                 const float* __restrict__ grow) {
    v16bf B;
    #pragma unroll
    for (int s = 0; s < 8; ++s) {
        const long o0 = (long)s * PIX, o1 = (long)(16 + s) * PIX;
        B[s]     = (__bf16)(brow[o0] * grow[o0]);
        B[8 + s] = (__bf16)(brow[o1] * grow[o1]);
    }
    return B;
}

// ---------------------------------------------------------------------------
// 1x1 conv as GEMM: Out[co, p] = act( sum_k W[co,k] * In[k,p] (* In2[k,p]) + b[co] )
// Wave tile: 32 Cout x 64 px, 8 accumulators, 8 WMMA per K-step of 32.
// Block = 128 threads = 4 waves (same Cout rows, adjacent px tiles).
// grid = (PIX/256, Cout/32, B).
// ---------------------------------------------------------------------------
template<int ACT, int PREMUL>
__global__ __launch_bounds__(128)
void conv1x1_wmma_kernel(const float* __restrict__ in,
                         const float* __restrict__ in2,
                         const float* __restrict__ Wt,
                         const float* __restrict__ bias,
                         float* __restrict__ out,
                         int Cin, long in_bstride, long out_bstride)
{
    const int lane  = threadIdx.x & 31;
    const int wave  = threadIdx.x >> 5;
    const int m     = lane & 15;       // A row within tile / C pixel-col
    const int half  = lane >> 4;       // lane half -> K sub-range / C row group
    const int pbase  = (blockIdx.x * 4 + wave) * 64;
    const int cobase = blockIdx.y * 32;

    const float* inb  = in + (long)blockIdx.z * in_bstride;
    const float* in2b = PREMUL ? (in2 + (long)blockIdx.z * in_bstride) : nullptr;
    float*       outb = out + (long)blockIdx.z * out_bstride;

    const float* wrow0 = Wt + (long)(cobase + m) * Cin + half * 8;
    const float* wrow1 = wrow0 + 16L * (long)Cin;

    v8f acc[2][4] = {};
    for (int kb = 0; kb < Cin; kb += 32) {
        const v16bf A0 = load_a_frag(wrow0 + kb);
        const v16bf A1 = load_a_frag(wrow1 + kb);

        const float* brow = inb + (long)(kb + half * 8) * PIX + pbase + m;
        __builtin_prefetch(brow + 32L * PIX, 0, 1);   // global_prefetch next K tile
        v16bf B[4];
        if (PREMUL) {
            const float* grow = in2b + (long)(kb + half * 8) * PIX + pbase + m;
            #pragma unroll
            for (int j = 0; j < 4; ++j) B[j] = load_b_frag_mul(brow + j*16, grow + j*16);
        } else {
            #pragma unroll
            for (int j = 0; j < 4; ++j) B[j] = load_b_frag(brow + j*16);
        }

        #pragma unroll
        for (int j = 0; j < 4; ++j) {
            acc[0][j] = __builtin_amdgcn_wmma_f32_16x16x32_bf16(
                            false, A0, false, B[j], (short)0, acc[0][j], false, false);
            acc[1][j] = __builtin_amdgcn_wmma_f32_16x16x32_bf16(
                            false, A1, false, B[j], (short)0, acc[1][j], false, false);
        }
    }

    // Epilogue: C layout co = cobase + i*16 + v + 8*half, px = pbase + j*16 + (lane&15)
    #pragma unroll
    for (int i = 0; i < 2; ++i) {
        #pragma unroll
        for (int v = 0; v < 8; ++v) {
            const int co = cobase + i * 16 + v + 8 * half;
            const float bv = bias[co];
            float* orow = outb + (long)co * PIX + pbase + m;
            #pragma unroll
            for (int j = 0; j < 4; ++j) {
                float val = acc[i][j][v] + bv;
                if (ACT) val = silu_f(val);
                orow[j * 16] = val;
            }
        }
    }
}

// ---------------------------------------------------------------------------
// TMaxAvgPool: 0.9*max5x5 + 0.1*(sum5x5/25).  One (b,c) plane per block,
// plane cached in LDS; bounds-checked window == (-inf pad for max, 0 for sum).
// ---------------------------------------------------------------------------
__global__ __launch_bounds__(256)
void tmaxavg_kernel(const float* __restrict__ in, float* __restrict__ out,
                    long bstride)
{
    __shared__ float tile[PIX];
    const int b  = blockIdx.x >> 8;
    const int cc = blockIdx.x & 255;
    const float* src = in  + (long)b * bstride + (long)cc * PIX;
    float*       dst = out + (long)b * bstride + (long)cc * PIX;

    for (int i = threadIdx.x; i < PIX; i += 256) tile[i] = src[i];
    __syncthreads();

    for (int i = threadIdx.x; i < PIX; i += 256) {
        const int x = i & 63, y = i >> 6;
        float mx = -__builtin_inff();
        float sm = 0.0f;
        #pragma unroll
        for (int dy = -2; dy <= 2; ++dy) {
            const int yy = y + dy;
            if ((unsigned)yy >= 64u) continue;
            #pragma unroll
            for (int dx = -2; dx <= 2; ++dx) {
                const int xx = x + dx;
                if ((unsigned)xx >= 64u) continue;
                const float v = tile[yy * HWD + xx];
                mx = fmaxf(mx, v);
                sm += v;
            }
        }
        dst[i] = TPOOL * mx + (1.0f - TPOOL) * (sm * (1.0f / 25.0f));
    }
}

// ---------------------------------------------------------------------------
// RWPool: sum(e*x)/(sum(e)+1e-6), e = exp(x - shift).  Shift-invariant ratio;
// per-window max used as the shift (numerically safe local softmax).
// ---------------------------------------------------------------------------
__global__ __launch_bounds__(256)
void rwpool_kernel(const float* __restrict__ in, float* __restrict__ out,
                   long bstride)
{
    __shared__ float tile[PIX];
    const int b  = blockIdx.x >> 8;
    const int cc = blockIdx.x & 255;
    const float* src = in  + (long)b * bstride + (long)cc * PIX;
    float*       dst = out + (long)b * bstride + (long)cc * PIX;

    for (int i = threadIdx.x; i < PIX; i += 256) tile[i] = src[i];
    __syncthreads();

    for (int i = threadIdx.x; i < PIX; i += 256) {
        const int x = i & 63, y = i >> 6;
        float mx = -__builtin_inff();
        #pragma unroll
        for (int dy = -2; dy <= 2; ++dy) {
            const int yy = y + dy;
            if ((unsigned)yy >= 64u) continue;
            #pragma unroll
            for (int dx = -2; dx <= 2; ++dx) {
                const int xx = x + dx;
                if ((unsigned)xx >= 64u) continue;
                mx = fmaxf(mx, tile[yy * HWD + xx]);
            }
        }
        float se = 0.0f, sev = 0.0f;
        #pragma unroll
        for (int dy = -2; dy <= 2; ++dy) {
            const int yy = y + dy;
            if ((unsigned)yy >= 64u) continue;
            #pragma unroll
            for (int dx = -2; dx <= 2; ++dx) {
                const int xx = x + dx;
                if ((unsigned)xx >= 64u) continue;
                const float v = tile[yy * HWD + xx];
                const float e = __expf(v - mx);
                se  += e;
                sev += e * v;
            }
        }
        dst[i] = sev / (se + 1e-6f);
    }
}

// ---------------------------------------------------------------------------
// Depthwise 3-tap conv (axis 0 = W, 1 = H) with dilation, zero padding.
// ---------------------------------------------------------------------------
__global__ __launch_bounds__(256)
void dwconv3_kernel(const float* __restrict__ in, const float* __restrict__ w,
                    const float* __restrict__ bias, float* __restrict__ out,
                    int axis, int dil, long total)
{
    const long idx = (long)blockIdx.x * 256 + threadIdx.x;
    if (idx >= total) return;
    const int x = (int)(idx & 63);
    const int y = (int)((idx >> 6) & 63);
    const int c = (int)((idx >> 12) & 1023);

    const float w0 = w[c * 3 + 0], w1 = w[c * 3 + 1], w2 = w[c * 3 + 2];
    float s = bias[c] + w1 * in[idx];
    if (axis == 0) {
        if (x - dil >= 0)  s += w0 * in[idx - dil];
        if (x + dil < HWD) s += w2 * in[idx + dil];
    } else {
        if (y - dil >= 0)  s += w0 * in[idx - (long)dil * HWD];
        if (y + dil < HWD) s += w2 * in[idx + (long)dil * HWD];
    }
    out[idx] = s;
}

// ---------------------------------------------------------------------------
extern "C" void kernel_launch(void* const* d_in, const int* in_sizes, int n_in,
                              void* d_out, int out_size, void* d_ws, size_t ws_size,
                              hipStream_t stream)
{
    (void)in_sizes; (void)n_in; (void)out_size; (void)ws_size;
    const float* x       = (const float*)d_in[0];
    const float* w_sta   = (const float*)d_in[1];  const float* b_sta   = (const float*)d_in[2];
    const float* w_cv1   = (const float*)d_in[3];  const float* b_cv1   = (const float*)d_in[4];
    const float* w_cv2   = (const float*)d_in[5];  const float* b_cv2   = (const float*)d_in[6];
    const float* w_cvend = (const float*)d_in[7];  const float* b_cvend = (const float*)d_in[8];
    const float* w_dwh   = (const float*)d_in[9];  const float* b_dwh   = (const float*)d_in[10];
    const float* w_dwv   = (const float*)d_in[11]; const float* b_dwv   = (const float*)d_in[12];
    const float* w_ddwh  = (const float*)d_in[13]; const float* b_ddwh  = (const float*)d_in[14];
    const float* w_ddwv  = (const float*)d_in[15]; const float* b_ddwv  = (const float*)d_in[16];
    const float* w_c1    = (const float*)d_in[17]; const float* b_c1    = (const float*)d_in[18];
    float* out = (float*)d_out;

    const long CAT_B = 1024L * PIX;          // per-batch floats, 1024-channel buffers
    const long SB    = 8L * CAT_B;           // floats per full [8,1024,64,64] buffer
    float* y1cat = (float*)d_ws;             // SPPF branch-1 concat (and later ping)
    float* y2cat = y1cat + SB;               // SPPF branch-2 concat (and later pong)
    float* ybuf  = y2cat + SB;               // y = cat(cv1, cv2)
    float* ping  = y1cat;
    float* pong  = y2cat;

    const dim3 cblk(128, 1, 1);
    auto cgrid = [](int Cout) { return dim3(PIX / 256, Cout / 32, 8); };

    // x_aux = CBS(x, w_sta) into slice 0 of both concat buffers
    conv1x1_wmma_kernel<1,0><<<cgrid(256), cblk, 0, stream>>>(
        x, nullptr, w_sta, b_sta, y1cat, 512, 512L * PIX, CAT_B);
    conv1x1_wmma_kernel<1,0><<<cgrid(256), cblk, 0, stream>>>(
        x, nullptr, w_sta, b_sta, y2cat, 512, 512L * PIX, CAT_B);

    // SPPF cascades: slice s-1 -> slice s
    const dim3 pgrid(8 * 256, 1, 1), pblk(256, 1, 1);
    for (int s = 1; s < 4; ++s) {
        tmaxavg_kernel<<<pgrid, pblk, 0, stream>>>(
            y1cat + (s - 1) * 256L * PIX, y1cat + s * 256L * PIX, CAT_B);
        rwpool_kernel<<<pgrid, pblk, 0, stream>>>(
            y2cat + (s - 1) * 256L * PIX, y2cat + s * 256L * PIX, CAT_B);
    }

    // y = cat( CBS(y1cat, w_cv1), CBS(y2cat, w_cv2) )
    conv1x1_wmma_kernel<1,0><<<cgrid(512), cblk, 0, stream>>>(
        y1cat, nullptr, w_cv1, b_cv1, ybuf, 1024, CAT_B, CAT_B);
    conv1x1_wmma_kernel<1,0><<<cgrid(512), cblk, 0, stream>>>(
        y2cat, nullptr, w_cv2, b_cv2, ybuf + 512L * PIX, 1024, CAT_B, CAT_B);

    // LSKA depthwise chain on ybuf (cat buffers recycled as ping/pong)
    const long total = 8L * 1024 * PIX;
    const int dwblocks = (int)(total / 256);
    dwconv3_kernel<<<dwblocks, 256, 0, stream>>>(ybuf, w_dwh,  b_dwh,  ping, 0, 1, total);
    dwconv3_kernel<<<dwblocks, 256, 0, stream>>>(ping, w_dwv,  b_dwv,  pong, 1, 1, total);
    dwconv3_kernel<<<dwblocks, 256, 0, stream>>>(pong, w_ddwh, b_ddwh, ping, 0, 2, total);
    dwconv3_kernel<<<dwblocks, 256, 0, stream>>>(ping, w_ddwv, b_ddwv, pong, 1, 2, total);

    // a = conv1x1(a, w_c1) (no activation)
    conv1x1_wmma_kernel<0,0><<<cgrid(1024), cblk, 0, stream>>>(
        pong, nullptr, w_c1, b_c1, ping, 1024, CAT_B, CAT_B);

    // out = CBS(y * a, w_cvend)  -- gating multiply fused into B-load path
    conv1x1_wmma_kernel<1,1><<<cgrid(512), cblk, 0, stream>>>(
        ybuf, ping, w_cvend, b_cvend, out, 1024, CAT_B, 512L * PIX);
}
